// ScaledDotProductAttention_77386720740036
// MI455X (gfx1250) — compile-verified, matrix operand order checked by reading, not run on hardware
//
#include <hip/hip_runtime.h>

typedef __bf16 bf16_t;
typedef bf16_t v16bf __attribute__((ext_vector_type(16)));
typedef float  v8f   __attribute__((ext_vector_type(8)));

#define B_   4
#define H_   16
#define S_   2048
#define D_   64
#define TQ   16
#define SCALE_F 0.125f           // 1/sqrt(64)
#define NEG_F  -1000000000.0f

// A/B fragment K-index mapping for 16-bit 16x32 WMMA operands (ISA 7.12.2):
// half j of the v16bf, lane l  ->  K = (j>=8 ? 16:0) + (l>=16 ? 8:0) + (j&7)
__device__ __forceinline__ int kmap(int j, int lane) {
    return ((j >= 8) ? 16 : 0) + ((lane >= 16) ? 8 : 0) + (j & 7);
}

extern "C" __global__ __launch_bounds__(256)
void sdpa_fused_kernel(const float* __restrict__ Q, const float* __restrict__ K,
                       const float* __restrict__ V, const int* __restrict__ Mask,
                       float* __restrict__ outO, float* __restrict__ outW)
{
    // Scores / exp buffer: rows padded (+4 dwords) so column-strided fragment
    // reads hit distinct banks (stride 2052 % 64 == 4) and rows stay 16B-aligned.
    __shared__ float  s_e[TQ][S_ + 4];          // 131328 B
    __shared__ bf16_t s_q[TQ][D_ + 8];          //   2304 B (stride 36 dwords)
    __shared__ float  s_out[2][TQ][D_];         //   8192 B
    __shared__ float  s_inv[TQ];

    const int tid  = threadIdx.x;
    const int lane = tid & 31;
    const int wid  = tid >> 5;

    const int wg = blockIdx.x;
    const int qt = wg & (S_ / TQ - 1);          // 128 q-tiles
    const int h  = (wg >> 7) & (H_ - 1);
    const int b  = wg >> 11;

    const size_t bh      = (size_t)(b * H_ + h);
    const size_t qkvBase = bh * (size_t)S_ * D_;
    const int    q0      = qt * TQ;
    const int    mrow    = lane & 15;

    // ---- Stage Q tile into LDS as bf16 (b128 global loads) ----
    for (int i = tid; i < TQ * D_ / 4; i += 256) {
        int r = i >> 4, c = (i & 15) * 4;
        float4 v = *(const float4*)&Q[qkvBase + (size_t)(q0 + r) * D_ + c];
        s_q[r][c + 0] = (bf16_t)v.x;
        s_q[r][c + 1] = (bf16_t)v.y;
        s_q[r][c + 2] = (bf16_t)v.z;
        s_q[r][c + 3] = (bf16_t)v.w;
    }
    __syncthreads();

    // ---- Per-wave Q A-fragments (reused for all 16 column tiles) ----
    v16bf aq0, aq1;
#pragma unroll
    for (int j = 0; j < 16; ++j) {
        int d = kmap(j, lane);
        aq0[j] = s_q[mrow][d];
        aq1[j] = s_q[mrow][d + 32];
    }

    // ---- QK^T, scale, mask -> LDS scores ----
    const size_t maskBase = ((size_t)b * S_ + q0) * S_;
    for (int ct = wid; ct < S_ / 16; ct += 8) {
        const int k0 = ct * 16;
        const float* Kp = K + qkvBase + (size_t)(k0 + mrow) * D_;
        v16bf bf0, bf1;
#pragma unroll
        for (int j = 0; j < 16; ++j) {
            int d = kmap(j, lane);
            bf0[j] = (bf16_t)Kp[d];
            bf1[j] = (bf16_t)Kp[d + 32];
        }
        v8f c = {};
        c = __builtin_amdgcn_wmma_f32_16x16x32_bf16(false, aq0, false, bf0,
                                                    (short)0, c, false, false);
        c = __builtin_amdgcn_wmma_f32_16x16x32_bf16(false, aq1, false, bf1,
                                                    (short)0, c, false, false);
        const int hi = (lane >> 4) << 3;        // lanes 16-31 hold rows M+8
#pragma unroll
        for (int r = 0; r < 8; ++r) {
            int m = r + hi;
            int n = k0 + mrow;
            float s = c[r] * SCALE_F;
            int mv  = Mask[maskBase + (size_t)m * S_ + n];
            s_e[m][n] = (mv == 0) ? NEG_F : s;
        }
    }
    __syncthreads();

    // ---- Row softmax (f32, float4 sweeps): 2 rows per wave ----
    for (int rr = 0; rr < 2; ++rr) {
        const int r = wid * 2 + rr;
        float4* row4 = (float4*)&s_e[r][0];     // row base is 16B-aligned

        float mx = -3.0e38f;
        for (int i = lane; i < S_ / 4; i += 32) {
            float4 v = row4[i];
            mx = fmaxf(mx, fmaxf(fmaxf(v.x, v.y), fmaxf(v.z, v.w)));
        }
#pragma unroll
        for (int off = 16; off > 0; off >>= 1)
            mx = fmaxf(mx, __shfl_xor(mx, off, 32));

        float sum = 0.0f;
        for (int i = lane; i < S_ / 4; i += 32) {
            float4 v = row4[i];
            float4 e;
            e.x = __expf(v.x - mx);
            e.y = __expf(v.y - mx);
            e.z = __expf(v.z - mx);
            e.w = __expf(v.w - mx);
            row4[i] = e;                        // keep un-normalized exp
            sum += (e.x + e.y) + (e.z + e.w);
        }
#pragma unroll
        for (int off = 16; off > 0; off >>= 1)
            sum += __shfl_xor(sum, off, 32);

        const float inv = 1.0f / sum;
        if (lane == 0) s_inv[r] = inv;

        // the ~1 GB weights stream: 128-bit coalesced stores
        float4* wrow4 = (float4*)(outW + (bh * (size_t)S_ + (size_t)(q0 + r)) * S_);
        for (int i = lane; i < S_ / 4; i += 32) {
            float4 e = row4[i];
            float4 w = make_float4(e.x * inv, e.y * inv, e.z * inv, e.w * inv);
            wrow4[i] = w;
        }
    }
    __syncthreads();

    // ---- P·V: 8 waves = 4 d-tiles x 2 k-halves; 1/sum folded at the end ----
    {
        const int dt = wid & 3;
        const int khalf = wid >> 2;
        const int d0 = dt * 16;
        const int kstart = khalf * (S_ / 2);
        v8f c = {};
        for (int k0 = kstart; k0 < kstart + S_ / 2; k0 += 32) {
            v16bf ap, bv;
#pragma unroll
            for (int j = 0; j < 16; ++j) {
                int kk = k0 + kmap(j, lane);
                ap[j] = (bf16_t)s_e[mrow][kk];
                bv[j] = (bf16_t)V[qkvBase + (size_t)kk * D_ + d0 + mrow];
            }
            c = __builtin_amdgcn_wmma_f32_16x16x32_bf16(false, ap, false, bv,
                                                        (short)0, c, false, false);
        }
        const int hi = (lane >> 4) << 3;
#pragma unroll
        for (int r = 0; r < 8; ++r)
            s_out[khalf][r + hi][d0 + mrow] = c[r];
    }
    __syncthreads();

    // ---- Combine k-halves, normalize, write output (one float4/thread) ----
    for (int i = tid; i < TQ * D_ / 4; i += 256) {
        int r = i >> 4, d = (i & 15) * 4;
        float4 a = *(float4*)&s_out[0][r][d];
        float4 bq = *(float4*)&s_out[1][r][d];
        float inv = s_inv[r];
        float4 o = make_float4((a.x + bq.x) * inv, (a.y + bq.y) * inv,
                               (a.z + bq.z) * inv, (a.w + bq.w) * inv);
        *(float4*)&outO[qkvBase + (size_t)(q0 + r) * D_ + d] = o;
    }
}

extern "C" void kernel_launch(void* const* d_in, const int* in_sizes, int n_in,
                              void* d_out, int out_size, void* d_ws, size_t ws_size,
                              hipStream_t stream) {
    (void)in_sizes; (void)n_in; (void)out_size; (void)d_ws; (void)ws_size;
    const float* Q   = (const float*)d_in[0];
    const float* K   = (const float*)d_in[1];
    const float* V   = (const float*)d_in[2];
    const int*   M   = (const int*)d_in[3];
    float* outO = (float*)d_out;                                  // [B,H,S,D]
    float* outW = outO + (size_t)B_ * H_ * S_ * D_;               // [B,H,S,S]

    dim3 grid(B_ * H_ * (S_ / TQ));   // 8192 workgroups
    hipLaunchKernelGGL(sdpa_fused_kernel, grid, dim3(256), 0, stream,
                       Q, K, V, M, outO, outW);
}
